// GUNet_4990751998612
// MI455X (gfx1250) — matrix operand order repaired and checked by reading, not compiled, not run on hardware
//
#include <hip/hip_runtime.h>
#include <hip/hip_bf16.h>

typedef __attribute__((ext_vector_type(16))) _Float16 v16h;
typedef __attribute__((ext_vector_type(8)))  _Float16 v8h;
typedef __attribute__((ext_vector_type(2)))  _Float16 v2h;
typedef __attribute__((ext_vector_type(8)))  float    v8f;

#define CDIV(a, b) (((a) + (b) - 1) / (b))
#define AL32(x) (((x) + 31) & ~31)

// ---------------------------------------------------------------------------
// Tiled WMMA GEMM: C[M,N](f32) = A[M,K](f16,lda) @ B[K,N](f16,ldb)
// A: rows 0..M-1 valid, cols padded/zero-filled to lda (multiple of 32);
//    OOB rows redirected to zrow (zero-filled, length >= lda).
// B: rows padded/zero-filled to >= ceil32(K), cols padded/zero-filled to ldb.
// 128x128x32 tiles, 256 threads = 8 waves (2x4), double-buffered LDS.
// A tile staged via GLOBAL_LOAD_ASYNC_TO_LDS_B128 (ASYNCcnt); B tile staged
// through VGPRs with in-register k-pair transpose. v_wmma_f32_16x16x32_f16.
// ---------------------------------------------------------------------------
__global__ __launch_bounds__(256)
void k_gemm_wmma(const _Float16* __restrict__ A, int lda,
                 const _Float16* __restrict__ B, int ldb,
                 const _Float16* __restrict__ zrow,
                 float* __restrict__ C, int M, int N, int K) {
    constexpr int BM = 128, BN = 128, BK = 32, LDW = 40;
    __shared__ alignas(16) _Float16 As[2][BM][LDW];
    __shared__ alignas(16) _Float16 Bs[2][BN][LDW];

    const int tid  = threadIdx.x;
    const int lane = tid & 31;
    const int wave = tid >> 5;
    const int wr   = wave >> 2;      // 0..1
    const int wc   = wave & 3;       // 0..3
    const int m0   = blockIdx.y * BM;
    const int n0   = blockIdx.x * BN;
    const int lm   = lane & 15;
    const int lh   = lane >> 4;

    // A staging: chunks tid and tid+256 of 512 (row = ch>>2, col8 = ch&3)
    const int ar0 = tid >> 2;
    const int ar1 = (tid + 256) >> 2;
    const int ac  = (tid & 3) * 8;
    // B staging: k-pair = 2*(tid&15), col chunk = (tid>>4)*8
    const int kp  = (tid & 15) * 2;
    const int bc  = (tid >> 4) * 8;

    const v8h vzero = {};

    const int gra0 = m0 + ar0, gra1 = m0 + ar1;
    const _Float16* pA0 = (gra0 < M) ? (A + (size_t)gra0 * lda + ac) : (zrow + ac);
    const _Float16* pA1 = (gra1 < M) ? (A + (size_t)gra1 * lda + ac) : (zrow + ac);
    int gcb = n0 + bc; const bool okB = gcb < ldb; if (!okB) gcb = ldb - 8;
    const _Float16* pB = B + (size_t)kp * ldb + gcb;

    v8h rB0, rB1;

    // async copy of this thread's two A chunks for tile at K0 into LDS[BUF]
#define ASYNC_A(K0, BUF)                                                  \
    do {                                                                  \
        unsigned l0 = (unsigned)(uintptr_t)&As[BUF][ar0][ac];             \
        unsigned l1 = (unsigned)(uintptr_t)&As[BUF][ar1][ac];             \
        asm volatile("global_load_async_to_lds_b128 %0, %1, off"          \
                     :: "v"(l0), "v"(pA0 + (K0)) : "memory");             \
        asm volatile("global_load_async_to_lds_b128 %0, %1, off"          \
                     :: "v"(l1), "v"(pA1 + (K0)) : "memory");             \
    } while (0)

#define WAIT_ASYNC() asm volatile("s_wait_asynccnt 0x0" ::: "memory")

#define LOAD_B(K0)                                                        \
    do {                                                                  \
        rB0 = *(const v8h*)(pB + (size_t)(K0) * ldb);                     \
        rB1 = *(const v8h*)(pB + (size_t)((K0) + 1) * ldb);               \
        if (!okB) { rB0 = vzero; rB1 = vzero; }                           \
    } while (0)

#define STORE_B(BUF)                                                      \
    do {                                                                  \
        _Pragma("unroll")                                                 \
        for (int j = 0; j < 8; ++j) {                                     \
            v2h pr; pr[0] = rB0[j]; pr[1] = rB1[j];                       \
            *(v2h*)&Bs[BUF][bc + j][kp] = pr;                             \
        }                                                                 \
    } while (0)

    v8f acc[4][2] = {};
    const int kTiles = (K + BK - 1) / BK;

    ASYNC_A(0, 0);
    LOAD_B(0);
    STORE_B(0);
    WAIT_ASYNC();
    __syncthreads();

    int cur = 0;
    for (int kt = 0; kt < kTiles; ++kt) {
        const bool hasNext = (kt + 1) < kTiles;
        if (hasNext) {
            ASYNC_A((kt + 1) * BK, cur ^ 1);   // DMA engine fills next A tile
            LOAD_B((kt + 1) * BK);             // overlapped with WMMAs below
        }

        v16h af[4], bf[2];
        #pragma unroll
        for (int t = 0; t < 4; ++t) {
            const _Float16* p = &As[cur][wr * 64 + t * 16 + lm][lh * 8];
            v8h lo = *(const v8h*)p;
            v8h hi = *(const v8h*)(p + 16);
            af[t] = __builtin_shufflevector(lo, hi, 0, 1, 2, 3, 4, 5, 6, 7,
                                            8, 9, 10, 11, 12, 13, 14, 15);
        }
        #pragma unroll
        for (int u = 0; u < 2; ++u) {
            const _Float16* p = &Bs[cur][wc * 32 + u * 16 + lm][lh * 8];
            v8h lo = *(const v8h*)p;
            v8h hi = *(const v8h*)(p + 16);
            bf[u] = __builtin_shufflevector(lo, hi, 0, 1, 2, 3, 4, 5, 6, 7,
                                            8, 9, 10, 11, 12, 13, 14, 15);
        }
        #pragma unroll
        for (int t = 0; t < 4; ++t)
            #pragma unroll
            for (int u = 0; u < 2; ++u)
                acc[t][u] = __builtin_amdgcn_wmma_f32_16x16x32_f16(
                    false, af[t], false, bf[u], (short)0, acc[t][u], false, false);

        if (hasNext) { STORE_B(cur ^ 1); }
        WAIT_ASYNC();
        __syncthreads();
        cur ^= 1;
    }
#undef ASYNC_A
#undef WAIT_ASYNC
#undef LOAD_B
#undef STORE_B

    #pragma unroll
    for (int t = 0; t < 4; ++t) {
        int rowBase = m0 + wr * 64 + t * 16 + lh * 8;
        #pragma unroll
        for (int u = 0; u < 2; ++u) {
            int col = n0 + wc * 32 + u * 16 + lm;
            if (col < N) {
                #pragma unroll
                for (int r = 0; r < 8; ++r) {
                    int row = rowBase + r;
                    if (row < M) C[(size_t)row * N + col] = acc[t][u][r];
                }
            }
        }
    }
}

// ---------------------------------------------------------------------------
// Converters to padded, zero-filled f16 operand buffers
// ---------------------------------------------------------------------------
__global__ void k_cvt_f16(const float* __restrict__ src, int rows, int cols,
                          _Float16* __restrict__ dst, int rp, int cp) {
    size_t i = (size_t)blockIdx.x * blockDim.x + threadIdx.x;
    if (i >= (size_t)rp * cp) return;
    int r = (int)(i / cp), c = (int)(i - (size_t)r * cp);
    float v = (r < rows && c < cols) ? src[(size_t)r * cols + c] : 0.0f;
    dst[i] = (_Float16)v;
}

// augment_adj step 1 fused with f16 conversion: dst = A with diag forced to 1
__global__ void k_selfloop_f16(const float* __restrict__ A, int n,
                               _Float16* __restrict__ dst, int np) {
    size_t i = (size_t)blockIdx.x * blockDim.x + threadIdx.x;
    if (i >= (size_t)np * np) return;
    int r = (int)(i / np), c = (int)(i - (size_t)r * np);
    float v = 0.0f;
    if (r < n && c < n) v = (r == c) ? 1.0f : A[(size_t)r * n + c];
    dst[i] = (_Float16)v;
}

// ---------------------------------------------------------------------------
// Degree / normalization (parallel two-stage reduction)
// ---------------------------------------------------------------------------
__global__ void k_deg_partial(const float* __restrict__ A, int n, int rpc,
                              float* __restrict__ deg) {
    int c = blockIdx.x * blockDim.x + threadIdx.x;
    if (c >= n) return;
    int r0 = blockIdx.y * rpc;
    int r1 = r0 + rpc; if (r1 > n) r1 = n;
    float s = 0.0f;
    for (int r = r0; r < r1; ++r) s += A[(size_t)r * n + c];
    atomicAdd(&deg[c], s);
}

__global__ void k_deg_finish(const float* __restrict__ A, const float* __restrict__ deg,
                             int n, float* __restrict__ dinv, float* __restrict__ fix) {
    int c = blockIdx.x * blockDim.x + threadIdx.x;
    if (c >= n) return;
    float diag = A[(size_t)c * n + c];
    float fx = (diag == 0.0f) ? 2.0f : 0.0f;
    float d = deg[c] + fx;
    dinv[c] = (d > 0.0f) ? (1.0f / sqrtf(d)) : 0.0f;
    fix[c] = fx;
}

// y = dinv ⊙ xw : f32 copy for epilogue + padded f16 copy for GEMM B operand
__global__ void k_rowscale2(const float* __restrict__ X, const float* __restrict__ dinv,
                            float* __restrict__ Yf, _Float16* __restrict__ Yh,
                            int n, int H, int rp, int cp) {
    size_t i = (size_t)blockIdx.x * blockDim.x + threadIdx.x;
    if (i >= (size_t)rp * cp) return;
    int r = (int)(i / cp), c = (int)(i - (size_t)r * cp);
    float v = 0.0f;
    if (r < n && c < H) {
        v = dinv[r] * X[(size_t)r * H + c];
        Yf[(size_t)r * H + c] = v;
    }
    Yh[i] = (_Float16)v;
}

// out[r,h] = dinv[r] * (T[r,h] + fix[r]*Y[r,h]) + b[h]  (optional relu)
__global__ void k_gcn_epilogue(const float* __restrict__ T, const float* __restrict__ Y,
                               const float* __restrict__ dinv, const float* __restrict__ fix,
                               const float* __restrict__ b, float* __restrict__ out,
                               int n, int H, int relu) {
    int i = blockIdx.x * blockDim.x + threadIdx.x;
    if (i >= n * H) return;
    int r = i / H, h = i - r * H;
    float v = dinv[r] * (T[i] + fix[r] * Y[i]) + b[h];
    out[i] = relu ? fmaxf(v, 0.0f) : v;
}

// ---------------------------------------------------------------------------
// Misc small kernels
// ---------------------------------------------------------------------------
__global__ void k_scatter_edges(const int* __restrict__ ei, int E2,
                                float* __restrict__ A, int N) {
    int e = blockIdx.x * blockDim.x + threadIdx.x;
    if (e < E2) {
        int s = ei[e], d = ei[E2 + e];
        A[(size_t)s * N + d] = 1.0f;
    }
}

__global__ void k_zero_diag(float* __restrict__ C, int n) {
    int i = blockIdx.x * blockDim.x + threadIdx.x;
    if (i < n) C[(size_t)i * n + i] = 0.0f;
}

__global__ void k_score(const float* __restrict__ X, const float* __restrict__ w,
                        float* __restrict__ s, int n, int H) {
    int i = blockIdx.x * blockDim.x + threadIdx.x;
    if (i >= n) return;
    float nrm = 0.0f, dot = 0.0f;
    for (int h = 0; h < H; ++h) { nrm += w[h] * w[h]; dot += X[(size_t)i * H + h] * w[h]; }
    s[i] = tanhf(dot / sqrtf(nrm));
}

#define SORT_SZ 8192
__global__ __launch_bounds__(1024)
void k_topk(const float* __restrict__ s, int n, int k, int* __restrict__ perm) {
    __shared__ float sc[SORT_SZ];
    __shared__ int   si[SORT_SZ];
    int tid = threadIdx.x;
    for (int i = tid; i < SORT_SZ; i += 1024) {
        sc[i] = (i < n) ? s[i] : -INFINITY;
        si[i] = i;
    }
    __syncthreads();
    for (int kk = 2; kk <= SORT_SZ; kk <<= 1) {
        for (int j = kk >> 1; j > 0; j >>= 1) {
            for (int i = tid; i < SORT_SZ; i += 1024) {
                int ixj = i ^ j;
                if (ixj > i) {
                    bool desc = ((i & kk) == 0);
                    float a = sc[i], b = sc[ixj];
                    bool sw = desc ? (a < b) : (a > b);
                    if (sw) {
                        sc[i] = b; sc[ixj] = a;
                        int t = si[i]; si[i] = si[ixj]; si[ixj] = t;
                    }
                }
            }
            __syncthreads();
        }
    }
    for (int t = tid; t < k; t += 1024) perm[t] = si[t];
}

__global__ void k_gather_x(const float* __restrict__ X, const float* __restrict__ s,
                           const int* __restrict__ perm, float* __restrict__ Xp,
                           int k, int H) {
    int i = blockIdx.x * blockDim.x + threadIdx.x;
    if (i >= k * H) return;
    int t = i / H, h = i - t * H;
    int p = perm[t];
    Xp[i] = X[(size_t)p * H + h] * s[p];
}

__global__ void k_gather_A(const float* __restrict__ A, const int* __restrict__ perm,
                           float* __restrict__ Asub, int k, int n) {
    size_t i = (size_t)blockIdx.x * blockDim.x + threadIdx.x;
    if (i >= (size_t)k * k) return;
    int r = (int)(i / k), c = (int)(i - (size_t)r * k);
    Asub[i] = A[(size_t)perm[r] * n + perm[c]];
}

__global__ void k_add_scatter(const float* __restrict__ xc, const int* __restrict__ perm,
                              float* __restrict__ buf, int k, int H) {
    int i = blockIdx.x * blockDim.x + threadIdx.x;
    if (i >= k * H) return;
    int t = i / H, h = i - t * H;
    buf[(size_t)perm[t] * H + h] += xc[i];
}

__global__ void k_logsoftmax(const float* __restrict__ X, float* __restrict__ out,
                             int n, int C) {
    int i = blockIdx.x * blockDim.x + threadIdx.x;
    if (i >= n) return;
    const float* row = X + (size_t)i * C;
    float m = -INFINITY;
    for (int j = 0; j < C; ++j) m = fmaxf(m, row[j]);
    float sum = 0.0f;
    for (int j = 0; j < C; ++j) sum += expf(row[j] - m);
    float l = logf(sum) + m;
    for (int j = 0; j < C; ++j) out[(size_t)i * C + j] = row[j] - l;
}

// ---------------------------------------------------------------------------
// Host-side orchestration
// ---------------------------------------------------------------------------
struct WSP {
    float *xw, *yb, *Tb, *deg, *dinv, *fix;
    _Float16 *Ah, *Xh, *Wh, *yh, *zr;
};

static inline void launch_gemm(const _Float16* A, int lda, const _Float16* B, int ldb,
                               const _Float16* zrow, float* C, int M, int N, int K,
                               hipStream_t st) {
    dim3 g(CDIV(N, 128), CDIV(M, 128));
    k_gemm_wmma<<<g, 256, 0, st>>>(A, lda, B, ldb, zrow, C, M, N, K);
}

// GCNConv(improved): out = dinv ⊙ (Ahat @ (dinv ⊙ (X@W))) + b (+relu)
static void run_gcn(const float* A, const float* X, const float* W, const float* bias,
                    float* out, int n, int hin, int hout, int relu,
                    const WSP& w, hipStream_t st) {
    const int np = AL32(n), hip_ = AL32(hin), hop = AL32(hout);
    // degrees
    hipMemsetAsync(w.deg, 0, n * sizeof(float), st);
    dim3 gdeg(CDIV(n, 256), 32);
    k_deg_partial<<<gdeg, 256, 0, st>>>(A, n, CDIV(n, 32), w.deg);
    k_deg_finish<<<CDIV(n, 256), 256, 0, st>>>(A, w.deg, n, w.dinv, w.fix);
    // xw = X @ W
    k_cvt_f16<<<CDIV((size_t)np * hip_, 256), 256, 0, st>>>(X, n, hin, w.Xh, np, hip_);
    k_cvt_f16<<<CDIV((size_t)hip_ * hop, 256), 256, 0, st>>>(W, hin, hout, w.Wh, hip_, hop);
    launch_gemm(w.Xh, hip_, w.Wh, hop, w.zr, w.xw, n, hout, hin, st);
    // y = dinv ⊙ xw (f32 + padded f16)
    k_rowscale2<<<CDIV((size_t)np * hop, 256), 256, 0, st>>>(w.xw, w.dinv, w.yb, w.yh,
                                                             n, hout, np, hop);
    // T = A @ y
    k_cvt_f16<<<CDIV((size_t)np * np, 256), 256, 0, st>>>(A, n, n, w.Ah, np, np);
    launch_gemm(w.Ah, np, w.yh, hop, w.zr, w.Tb, n, hout, n, st);
    k_gcn_epilogue<<<CDIV(n * hout, 256), 256, 0, st>>>(w.Tb, w.yb, w.dinv, w.fix,
                                                        bias, out, n, hout, relu);
}

extern "C" void kernel_launch(void* const* d_in, const int* in_sizes, int n_in,
                              void* d_out, int out_size, void* d_ws, size_t ws_size,
                              hipStream_t stream) {
    const float* x   = (const float*)d_in[0];
    const int*   ei  = (const int*)d_in[1];
    const float* Wd0 = (const float*)d_in[2];  const float* bd0 = (const float*)d_in[3];
    const float* Wd1 = (const float*)d_in[4];  const float* bd1 = (const float*)d_in[5];
    const float* Wd2 = (const float*)d_in[6];  const float* bd2 = (const float*)d_in[7];
    const float* Wd3 = (const float*)d_in[8];  const float* bd3 = (const float*)d_in[9];
    const float* pw1 = (const float*)d_in[10];
    const float* pw2 = (const float*)d_in[11];
    const float* pw3 = (const float*)d_in[12];
    const float* Wu1 = (const float*)d_in[13]; const float* bu1 = (const float*)d_in[14];
    const float* Wu2 = (const float*)d_in[15]; const float* bu2 = (const float*)d_in[16];
    const float* Wu3 = (const float*)d_in[17]; const float* bu3 = (const float*)d_in[18];
    float* out = (float*)d_out;

    const int N = 5000, CIN = 32, H = 64, COUT = 8;
    const int K1 = 2000, K2 = 1000, K3 = 500;
    const int NP = AL32(N);
    const int E2 = in_sizes[1] / 2;

    char* base = (char*)d_ws;
    size_t off = 0;
    auto allocB = [&](size_t bytes) -> void* {
        void* p = (void*)(base + off);
        off += (bytes + 255) & ~(size_t)255;
        return p;
    };
    float* A0  = (float*)allocB((size_t)N * N * 4);      // As[0]
    float* Cb  = (float*)allocB((size_t)N * N * 4);      // augment result (f32)
    float* A1  = (float*)allocB((size_t)K1 * K1 * 4);    // As[1]
    float* A2  = (float*)allocB((size_t)K2 * K2 * 4);    // As[2]
    float* A3  = (float*)allocB((size_t)K3 * K3 * 4);    // deepest adjacency
    WSP w;
    w.Ah = (_Float16*)allocB((size_t)NP * NP * 2);       // f16 operand (adjacency)
    w.Xh = (_Float16*)allocB((size_t)NP * H * 2);
    w.Wh = (_Float16*)allocB((size_t)H * H * 2);
    w.yh = (_Float16*)allocB((size_t)NP * H * 2);
    w.zr = (_Float16*)allocB((size_t)NP * 2);            // zero row for OOB async A
    w.xw = (float*)allocB((size_t)N * H * 4);
    w.yb = (float*)allocB((size_t)N * H * 4);
    w.Tb = (float*)allocB((size_t)N * H * 4);
    w.deg  = (float*)allocB((size_t)N * 4);
    w.dinv = (float*)allocB((size_t)N * 4);
    w.fix  = (float*)allocB((size_t)N * 4);
    float* xs0 = (float*)allocB((size_t)N * H * 4);
    float* xs1 = (float*)allocB((size_t)K1 * H * 4);
    float* xs2 = (float*)allocB((size_t)K2 * H * 4);
    float* xg  = (float*)allocB((size_t)N * H * 4);
    float* xcb = (float*)allocB((size_t)N * H * 4);
    float* xin = (float*)allocB((size_t)N * H * 4);
    float* sbuf = (float*)allocB((size_t)N * 4);
    int* perm1 = (int*)allocB((size_t)K1 * 4);
    int* perm2 = (int*)allocB((size_t)K2 * 4);
    int* perm3 = (int*)allocB((size_t)K3 * 4);
    (void)ws_size; (void)n_in; (void)out_size;

    // ---- build dense adjacency + zero row ----
    hipMemsetAsync(A0, 0, (size_t)N * N * sizeof(float), stream);
    hipMemsetAsync(w.zr, 0, (size_t)NP * sizeof(_Float16), stream);
    k_scatter_edges<<<CDIV(E2, 256), 256, 0, stream>>>(ei, E2, A0, N);

    // ---- initial GCN ----
    run_gcn(A0, x, Wd0, bd0, xs0, N, CIN, H, 1, w, stream);

    struct Lvl { const float* Ain; const float* Xin; const float* pw;
                 const float* W; const float* b; float* Aout; float* Xout;
                 int n; int k; int* perm; };
    Lvl lv[3] = {
        { A0, xs0, pw1, Wd1, bd1, A1, xs1, N,  K1, perm1 },
        { A1, xs1, pw2, Wd2, bd2, A2, xs2, K1, K2, perm2 },
        { A2, xs2, pw3, Wd3, bd3, A3, xcb, K2, K3, perm3 },
    };

    // ---- down path: augment -> top-k pool -> GCN ----
    for (int i = 0; i < 3; ++i) {
        int n = lv[i].n, k = lv[i].k, np = AL32(n);
        k_selfloop_f16<<<CDIV((size_t)np * np, 256), 256, 0, stream>>>(lv[i].Ain, n,
                                                                       w.Ah, np);
        launch_gemm(w.Ah, np, w.Ah, np, w.zr, Cb, n, n, n, stream);  // A1 @ A1 (WMMA)
        k_zero_diag<<<CDIV(n, 256), 256, 0, stream>>>(Cb, n);
        k_score<<<CDIV(n, 256), 256, 0, stream>>>(lv[i].Xin, lv[i].pw, sbuf, n, H);
        k_topk<<<1, 1024, 0, stream>>>(sbuf, n, k, lv[i].perm);
        k_gather_x<<<CDIV(k * H, 256), 256, 0, stream>>>(lv[i].Xin, sbuf, lv[i].perm,
                                                         xg, k, H);
        k_gather_A<<<CDIV((size_t)k * k, 256), 256, 0, stream>>>(Cb, lv[i].perm,
                                                                 lv[i].Aout, k, n);
        run_gcn(lv[i].Aout, xg, lv[i].W, lv[i].b, lv[i].Xout, k, H, H, 1, w, stream);
    }

    // ---- up path: unpool (sum_res) -> GCN ----
    hipMemcpyAsync(xin, xs2, (size_t)K2 * H * sizeof(float),
                   hipMemcpyDeviceToDevice, stream);
    k_add_scatter<<<CDIV(K3 * H, 256), 256, 0, stream>>>(xcb, perm3, xin, K3, H);
    run_gcn(A2, xin, Wu1, bu1, xcb, K2, H, H, 1, w, stream);

    hipMemcpyAsync(xin, xs1, (size_t)K1 * H * sizeof(float),
                   hipMemcpyDeviceToDevice, stream);
    k_add_scatter<<<CDIV(K2 * H, 256), 256, 0, stream>>>(xcb, perm2, xin, K2, H);
    run_gcn(A1, xin, Wu2, bu2, xcb, K1, H, H, 1, w, stream);

    hipMemcpyAsync(xin, xs0, (size_t)N * H * sizeof(float),
                   hipMemcpyDeviceToDevice, stream);
    k_add_scatter<<<CDIV(K1 * H, 256), 256, 0, stream>>>(xcb, perm1, xin, K1, H);
    run_gcn(A0, xin, Wu3, bu3, xg, N, H, COUT, 0, w, stream);

    // ---- log_softmax ----
    k_logsoftmax<<<CDIV(N, 256), 256, 0, stream>>>(xg, out, N, COUT);
}
